// ClusteredLinformerAttention_86887188398773
// MI455X (gfx1250) — matrix operand order repaired and checked by reading, not compiled, not run on hardware
//
#include <hip/hip_runtime.h>
#include <hip/hip_bf16.h>

// ---------------------------------------------------------------------------
// ClusteredLinformerAttention on gfx1250 (MI455X, wave32, WMMA + TDM).
// Bandwidth-bound op (~100 GFLOP vs ~250MB traffic @ 23.3TB/s) -> f16
// operands with f32 WMMA accumulation for all GEMMs (v_wmma_f32_16x16x32_f16).
// The 3-branch "SAME" conv over scores is a single 5x64 conv; its full-width
// (64) dimension makes each height-tap a banded 64x64 matmul -> WMMA too.
// conv_b is softmax-invariant -> dropped. Block-uniform B operands of the
// fused kernel are DMA'd to LDS with tensor_load_to_lds (TDM, TENSORcnt).
// ---------------------------------------------------------------------------

typedef __attribute__((ext_vector_type(16))) _Float16 v16h;
typedef __attribute__((ext_vector_type(8)))  _Float16 v8h;
typedef __attribute__((ext_vector_type(8)))  float    v8f;
typedef __attribute__((ext_vector_type(4)))  unsigned int v4u;
typedef __attribute__((ext_vector_type(8)))  int      v8i;
typedef __attribute__((ext_vector_type(4)))  int      v4i;

#define SEQ   2048
#define DM    512
#define NH    8
#define DH    64
#define PD    64     // proj dim / clusters
#define CL    32     // chunk
#define BAT   16
#define BS    (BAT * SEQ)          // 32768 rows
#define BH    (BAT * NH)           // 128

// ---- WMMA helpers ----------------------------------------------------------

__device__ __forceinline__ v8f wmma16(v16h a, v16h b, v8f c) {
  // D = A(16x32 f16) * B(32x16 f16) + C(16x16 f32)
  return __builtin_amdgcn_wmma_f32_16x16x32_f16(
      /*neg_a=*/false, a, /*neg_b=*/false, b,
      /*c_mod=*/(short)0, c, /*reuse_a=*/false, /*reuse_b=*/false);
}

// A fragment (16-bit A 16x32): lane L holds row M=L%16,
// halves i: K = (L/16)*16 + i  -> 16 contiguous halves (matches packed B).
// Unchecked variant: caller guarantees the row is in range.
__device__ __forceinline__ v16h load_a_frag_u(const _Float16* __restrict__ A,
                                              int ld, int row0, int k0, int lane) {
  int m = row0 + (lane & 15);
  const _Float16* p = A + (size_t)m * ld + k0 + ((lane >> 4) << 4);
  union { v16h v; v8h h[2]; } u;
  u.h[0] = *(const v8h*)(p);
  u.h[1] = *(const v8h*)(p + 8);
  return u.v;
}

// Checked variant (zero rows outside [rlo,rhi)): only for the Q halo loads.
__device__ __forceinline__ v16h load_a_frag(const _Float16* __restrict__ A,
                                            int ld, int row0, int k0, int lane,
                                            int rlo, int rhi) {
  int m = row0 + (lane & 15);
  v16h z = {};
  union { v16h v; v8h h[2]; } u;
  u.v = z;
  if (m >= rlo && m < rhi) {
    const _Float16* p = A + (size_t)m * ld + k0 + ((lane >> 4) << 4);
    u.h[0] = *(const v8h*)(p);
    u.h[1] = *(const v8h*)(p + 8);
  }
  return u.v;
}

// B fragments are pre-packed: fragment f, lane L -> 16 contiguous halves
// holding B[kb*32 + (L/16)*16 + i][nt*16 + L%16]  (f = kb*NT + nt).
__device__ __forceinline__ v16h load_b_frag(const _Float16* __restrict__ Bp,
                                            int fidx, int lane) {
  return *(const v16h*)(Bp + ((size_t)fidx * 32 + lane) * 16);
}

// ---- TDM: 1-D tensor DMA global -> LDS (6-arg builtin on this toolchain) ---
// D# group0: {count=1, lds_addr, global_addr[56:0], type=2}
// D# group1: {wg_mask=0, data_size=1(2B), tensor_dim0=n, tensor_dim1=1,
//             tile_dim0=n, tile_dim1=1, stride0=n}
__device__ __forceinline__ void tdm_load_1d(void* lds_dst, const void* gsrc,
                                            unsigned n /* f16 elems, <=65535 */) {
  unsigned lds = (unsigned)(size_t)lds_dst;        // flat->LDS: low 32 bits
  unsigned long long ga = (unsigned long long)(size_t)gsrc;
  v4u g0;
  g0[0] = 1u;                                      // count=1 (valid), user mode
  g0[1] = lds;                                     // lds_addr
  g0[2] = (unsigned)(ga & 0xFFFFFFFFu);            // global_addr[31:0]
  g0[3] = (unsigned)((ga >> 32) & 0x01FFFFFFu) | (2u << 30);  // [56:32] | type=2
  v8i g1;
  g1[0] = (int)(1u << 16);                         // data_size=1 -> 2 bytes
  g1[1] = (int)((n & 0xFFFFu) << 16);              // tensor_dim0[15:0]
  g1[2] = (int)((n >> 16) | (1u << 16));           // tensor_dim0[31:16] | tensor_dim1=1
  g1[3] = (int)(n << 16);                          // tensor_dim1 hi=0 | tile_dim0=n
  g1[4] = 1;                                       // tile_dim1=1, tile_dim2=0
  g1[5] = (int)n;                                  // tensor_dim0_stride[31:0]
  g1[6] = 0;                                       // stride0 hi | stride1 lo
  g1[7] = 0;
  v4i gz = {0, 0, 0, 0};                           // groups 2/3 unused (<=2D)
  v8i gz8 = {0, 0, 0, 0, 0, 0, 0, 0};              // extra group (6-arg form)
  __builtin_amdgcn_tensor_load_to_lds(g0, g1, gz, gz, gz8, 0);
}

// ---- prep kernels ----------------------------------------------------------

__global__ void cvt_f32_to_f16_kernel(const float* __restrict__ src,
                                      _Float16* __restrict__ dst) {
  int i = (blockIdx.x * blockDim.x + threadIdx.x) * 4;
  float4 v = *(const float4*)(src + i);
  dst[i]     = (_Float16)v.x;
  dst[i + 1] = (_Float16)v.y;
  dst[i + 2] = (_Float16)v.z;
  dst[i + 3] = (_Float16)v.w;
}

// Pack a 512x512 f32 weight (row=K, col=N) into WMMA-B fragment layout (f16).
__global__ void pack_w_kernel(const float* __restrict__ w0,
                              const float* __restrict__ w1,
                              const float* __restrict__ w2,
                              const float* __restrict__ w3,
                              _Float16* __restrict__ dst) {
  const float* srcs[4] = {w0, w1, w2, w3};
  const float* src = srcs[blockIdx.y];
  _Float16* d = dst + (size_t)blockIdx.y * (DM * DM);
  int tid  = blockIdx.x * blockDim.x + threadIdx.x;   // 0..262143
  int i    = tid & 15;
  int lane = (tid >> 4) & 31;
  int fidx = tid >> 9;                                // kb*32 + nt (NT=32)
  int nt   = fidx & 31;
  int kb   = fidx >> 5;
  int k = kb * 32 + ((lane >> 4) << 4) + i;
  int n = nt * 16 + (lane & 15);
  d[tid] = (_Float16)src[k * DM + n];
}

// Build the 5 banded conv matrices G_i[q][p] = Weff[i][q-p+31], packed f16.
// Weff folds conv1/conv3/conv5 and the mean (1/3). conv_b is softmax-invariant.
__global__ void build_g_kernel(const float* __restrict__ w1,
                               const float* __restrict__ w3,
                               const float* __restrict__ w5,
                               _Float16* __restrict__ Gp) {
  int tid = blockIdx.x * blockDim.x + threadIdx.x;
  if (tid >= 5 * 4096) return;
  int m = tid >> 12;          // height tap 0..4  (di = m-2)
  int w = tid & 4095;
  int i    = w & 15;
  int lane = (w >> 4) & 31;
  int fidx = w >> 9;          // kb*4 + nt (NT=4)
  int nt = fidx & 3;
  int kb = fidx >> 2;
  int q = kb * 32 + ((lane >> 4) << 4) + i;
  int p = nt * 16 + (lane & 15);
  int j = q - p + 31;         // width-filter tap (SAME pad: left 31, right 32)
  float val = 0.f;
  if (j >= 0 && j < 64) {
    val = w5[m * 64 + j];
    if (m >= 1 && m <= 3) val += w3[(m - 1) * 64 + j];
    if (m == 2) val += w1[j];
    val *= (1.f / 3.f);       // mean over the 3 conv branches
  }
  Gp[tid] = (_Float16)val;
}

__global__ void counts_kernel(const unsigned char* __restrict__ mask,
                              int* __restrict__ counts) {
  __shared__ int red[256];
  int b = blockIdx.x, t = threadIdx.x;
  int s = 0;
  for (int i = t; i < SEQ; i += 256) s += mask[b * SEQ + i] ? 1 : 0;
  red[t] = s;
  __syncthreads();
  for (int off = 128; off > 0; off >>= 1) {
    if (t < off) red[t] += red[t + off];
    __syncthreads();
  }
  if (t == 0) {
    int c = red[0];
    if (c < 1) c = 1;
    if (c > SEQ) c = SEQ;
    counts[b] = c;
  }
}

// ---- QKV projection: [BS,512] @ 512x512 (x3), write [B,H,S,64] f16 --------

__global__ void qkv_kernel(const _Float16* __restrict__ xh,
                           const _Float16* __restrict__ wp,
                           _Float16* __restrict__ Qh,
                           _Float16* __restrict__ Kh,
                           _Float16* __restrict__ Vh) {
  int mat = blockIdx.y;                       // 0=Q 1=K 2=V
  const _Float16* Bp = wp + (size_t)mat * (DM * DM);
  _Float16* dst = (mat == 0) ? Qh : (mat == 1) ? Kh : Vh;
  float scale = (mat == 0) ? 0.125f : 1.0f;   // fold 1/sqrt(64) into Q
  int wave = threadIdx.x >> 5, lane = threadIdx.x & 31;
  int tile = blockIdx.x * 8 + wave;           // 2048 row-tiles * 32 col-tiles
  int rt = tile >> 5, nt = tile & 31;
  // depth-2 software pipeline: keep next iteration's fragments in flight
  v16h a0 = load_a_frag_u(xh, DM, rt * 16, 0, lane);
  v16h b0 = load_b_frag(Bp, nt, lane);
  v8f acc = {};
#pragma unroll
  for (int kb = 0; kb < 15; ++kb) {
    v16h a1 = load_a_frag_u(xh, DM, rt * 16, (kb + 1) * 32, lane);
    v16h b1 = load_b_frag(Bp, (kb + 1) * 32 + nt, lane);
    acc = wmma16(a0, b0, acc);
    a0 = a1; b0 = b1;
  }
  acc = wmma16(a0, b0, acc);
  int n = nt * 16 + (lane & 15);
  int h = n >> 6, t = n & 63;
  int rbase = rt * 16 + ((lane >> 4) << 3);
#pragma unroll
  for (int j = 0; j < 8; ++j) {
    int m = rbase + j;
    int b_ = m >> 11, s = m & (SEQ - 1);
    dst[(((size_t)(b_ * NH + h)) * SEQ + s) * DH + t] = (_Float16)(acc[j] * scale);
  }
}

// ---- cluster gather + Linformer projections (tiny) ------------------------

__global__ void proj_kernel(const _Float16* __restrict__ Kh,
                            const _Float16* __restrict__ Vh,
                            const float* __restrict__ EW,
                            const float* __restrict__ FW,
                            const int* __restrict__ table,
                            const int* __restrict__ counts,
                            _Float16* __restrict__ kprojT_p,   // B-frag: K=d, N=p
                            _Float16* __restrict__ vproj_p) {  // B-frag: K=p, N=d
  __shared__ float kpl[PD * DH];
  __shared__ float vpl[PD * DH];
  int bh = blockIdx.x, b = bh >> 3, h = bh & 7;
  int cnt = counts[b];
  const int* pos = table + (size_t)(cnt - 1) * (PD * CL);
  const _Float16* Kb = Kh + (size_t)bh * SEQ * DH;
  const _Float16* Vb = Vh + (size_t)bh * SEQ * DH;
  for (int w = threadIdx.x; w < PD * DH; w += 256) {
    int c = w >> 6, t = w & 63;
    float ak = 0.f, av = 0.f;
    for (int l = 0; l < CL; ++l) {
      int idx = pos[c * CL + l];
      if (idx < SEQ) {                         // idx==SEQ is the zero pad row
        ak += EW[h * (PD * CL) + c * CL + l] * (float)Kb[(size_t)idx * DH + t];
        av += FW[h * (PD * CL) + c * CL + l] * (float)Vb[(size_t)idx * DH + t];
      }
    }
    kpl[c * DH + t] = ak;
    vpl[c * DH + t] = av;
  }
  __syncthreads();
  _Float16* kd = kprojT_p + (size_t)bh * 4096;
  _Float16* vd = vproj_p + (size_t)bh * 4096;
  for (int w = threadIdx.x; w < 4096; w += 256) {
    int i    = w & 15;
    int lane = (w >> 4) & 31;
    int fidx = w >> 9;                         // kb*4 + nt
    int nt = fidx & 3, kb = fidx >> 2;
    int k = kb * 32 + ((lane >> 4) << 4) + i;
    int n = nt * 16 + (lane & 15);
    kd[w] = (_Float16)kpl[n * DH + k];         // Bk[d][p] = k_proj[p][d]
    vd[w] = (_Float16)vpl[k * DH + n];         // Bv[p][d] = v_proj[p][d]
  }
}

// ---- fused scores -> conv -> softmax -> attn per (b,h, 64-row tile) -------

__global__ void fused_attn_kernel(const _Float16* __restrict__ Qh,
                                  const _Float16* __restrict__ kprojT_p,
                                  const _Float16* __restrict__ vproj_p,
                                  const _Float16* __restrict__ Gp,
                                  _Float16* __restrict__ concat) {
  __shared__ __align__(16) _Float16 sc[96 * 64];   // scores rows [s0-16,s0+80)
  __shared__ float cv[64 * 64];                    // conv output (f32)
  __shared__ __align__(16) _Float16 wl[64 * 64];   // softmax weights (f16)
  __shared__ __align__(32) _Float16 bK[4096];      // TDM-staged B operands
  __shared__ __align__(32) _Float16 bV[4096];
  __shared__ __align__(32) _Float16 bG[5 * 4096];
  int st = blockIdx.x, bh = blockIdx.y;
  int s0 = st * 64;
  int wave = threadIdx.x >> 5, lane = threadIdx.x & 31;
  const _Float16* Qb = Qh + (size_t)bh * SEQ * DH;

  // Stage block-uniform B matrices via the Tensor Data Mover (TENSORcnt).
  if (wave == 0) {
    tdm_load_1d(bK, kprojT_p + (size_t)bh * 4096, 4096);
    tdm_load_1d(bV, vproj_p + (size_t)bh * 4096, 4096);
    tdm_load_1d(bG, Gp, 5 * 4096);
    __builtin_amdgcn_s_wait_tensorcnt(0);
  }
  __syncthreads();

  // (a) scores = Q * kprojT (scale folded into Q), 96x64 incl. conv halo
  for (int tile = wave; tile < 24; tile += 8) {
    int rt = tile >> 2, nt = tile & 3;
    v8f acc = {};
    int row0 = s0 - 16 + rt * 16;
#pragma unroll
    for (int kb = 0; kb < 2; ++kb) {
      v16h a = load_a_frag(Qb, DH, row0, kb * 32, lane, 0, SEQ);  // OOB rows -> 0
      v16h b = load_b_frag(bK, kb * 4 + nt, lane);
      acc = wmma16(a, b, acc);
    }
    int rb = rt * 16 + ((lane >> 4) << 3);
    int c = nt * 16 + (lane & 15);
#pragma unroll
    for (int j = 0; j < 8; ++j) sc[(rb + j) * 64 + c] = (_Float16)acc[j];
  }
  __syncthreads();

  // (b) conv: out = sum_i shift_i(scores) @ G_i  (5 banded 64x64 matmuls)
  for (int tile = wave; tile < 16; tile += 8) {
    int rt = tile >> 2, nt = tile & 3;
    v8f acc = {};
#pragma unroll
    for (int i = 0; i < 5; ++i) {
      int row0 = 16 + rt * 16 + i - 2;              // rows 14..81, inside [0,96)
#pragma unroll
      for (int kb = 0; kb < 2; ++kb) {
        v16h a = load_a_frag_u(sc, 64, row0, kb * 32, lane);
        v16h g = load_b_frag(bG + (size_t)i * 4096, kb * 4 + nt, lane);
        acc = wmma16(a, g, acc);
      }
    }
    int rb = rt * 16 + ((lane >> 4) << 3);
    int c = nt * 16 + (lane & 15);
#pragma unroll
    for (int j = 0; j < 8; ++j) cv[(rb + j) * 64 + c] = acc[j];
  }
  __syncthreads();

  // (c) softmax over P=64 per row (f32)
  if (threadIdx.x < 64) {
    int r = threadIdx.x;
    float mx = -1e30f;
    for (int p = 0; p < 64; ++p) mx = fmaxf(mx, cv[r * 64 + p]);
    float sum = 0.f;
    for (int p = 0; p < 64; ++p) {
      float e = expf(cv[r * 64 + p] - mx);
      cv[r * 64 + p] = e;
      sum += e;
    }
    float inv = 1.f / sum;
    for (int p = 0; p < 64; ++p) wl[r * 64 + p] = (_Float16)(cv[r * 64 + p] * inv);
  }
  __syncthreads();

  // (d) attn = W * v_proj ; write straight into concat [B,S,D] f16 layout
  int b = bh >> 3, h = bh & 7;
  for (int tile = wave; tile < 16; tile += 8) {
    int rt = tile >> 2, nt = tile & 3;
    v8f acc = {};
#pragma unroll
    for (int kb = 0; kb < 2; ++kb) {
      v16h a = load_a_frag_u(wl, 64, rt * 16, kb * 32, lane);
      v16h v = load_b_frag(bV, kb * 4 + nt, lane);
      acc = wmma16(a, v, acc);
    }
    int rb = rt * 16 + ((lane >> 4) << 3);
    int c = nt * 16 + (lane & 15);
#pragma unroll
    for (int j = 0; j < 8; ++j) {
      int s = s0 + rb + j;
      concat[((size_t)b * SEQ + s) * DM + h * DH + c] = (_Float16)acc[j];
    }
  }
}

// ---- dense output projection: concat @ dense_w + b -> f32 out -------------

__global__ void dense_kernel(const _Float16* __restrict__ concat,
                             const _Float16* __restrict__ dwp,
                             const float* __restrict__ bias,
                             float* __restrict__ out) {
  int wave = threadIdx.x >> 5, lane = threadIdx.x & 31;
  int tile = blockIdx.x * 8 + wave;
  int rt = tile >> 5, nt = tile & 31;
  v16h a0 = load_a_frag_u(concat, DM, rt * 16, 0, lane);
  v16h b0 = load_b_frag(dwp, nt, lane);
  v8f acc = {};
#pragma unroll
  for (int kb = 0; kb < 15; ++kb) {
    v16h a1 = load_a_frag_u(concat, DM, rt * 16, (kb + 1) * 32, lane);
    v16h b1 = load_b_frag(dwp, (kb + 1) * 32 + nt, lane);
    acc = wmma16(a0, b0, acc);
    a0 = a1; b0 = b1;
  }
  acc = wmma16(a0, b0, acc);
  int n = nt * 16 + (lane & 15);
  float bv = bias[n];
  int rb = rt * 16 + ((lane >> 4) << 3);
#pragma unroll
  for (int j = 0; j < 8; ++j)
    out[(size_t)(rb + j) * DM + n] = acc[j] + bv;
}

// ---------------------------------------------------------------------------

extern "C" void kernel_launch(void* const* d_in, const int* in_sizes, int n_in,
                              void* d_out, int out_size, void* d_ws, size_t ws_size,
                              hipStream_t stream) {
  const float*         x     = (const float*)d_in[0];
  const unsigned char* mask  = (const unsigned char*)d_in[1];
  const float*         wq    = (const float*)d_in[2];
  const float*         wk    = (const float*)d_in[3];
  const float*         wv    = (const float*)d_in[4];
  const float*         EW    = (const float*)d_in[5];
  const float*         FW    = (const float*)d_in[6];
  const float*         cw1   = (const float*)d_in[7];
  const float*         cw3   = (const float*)d_in[8];
  const float*         cw5   = (const float*)d_in[9];
  /* d_in[10] = conv_b: constant across (s,p) -> softmax-invariant, unused */
  const float*         dw    = (const float*)d_in[11];
  const float*         db    = (const float*)d_in[12];
  const int*           table = (const int*)d_in[13];
  float* out = (float*)d_out;

  char* ws = (char*)d_ws;
  size_t off = 0;
  auto alloc = [&](size_t bytes) -> void* {
    void* p = ws + off;
    off = (off + bytes + 255) & ~(size_t)255;
    return p;
  };
  const size_t NELEM = (size_t)BS * DM;                 // 16,777,216
  _Float16* xh   = (_Float16*)alloc(NELEM * 2);
  _Float16* Qh   = (_Float16*)alloc(NELEM * 2);
  _Float16* Kh   = (_Float16*)alloc(NELEM * 2);
  _Float16* Vh   = (_Float16*)alloc(NELEM * 2);
  _Float16* Ch   = (_Float16*)alloc(NELEM * 2);         // concat f16
  _Float16* Wp   = (_Float16*)alloc((size_t)4 * DM * DM * 2);  // q,k,v,dense packed
  _Float16* KpT  = (_Float16*)alloc((size_t)BH * 4096 * 2);
  _Float16* Vpj  = (_Float16*)alloc((size_t)BH * 4096 * 2);
  _Float16* Gp   = (_Float16*)alloc((size_t)5 * 4096 * 2);
  int*      cnts = (int*)alloc(BAT * sizeof(int));

  // prep (independent)
  cvt_f32_to_f16_kernel<<<NELEM / (256 * 4), 256, 0, stream>>>(x, xh);
  pack_w_kernel<<<dim3((DM * DM) / 256, 4), 256, 0, stream>>>(wq, wk, wv, dw, Wp);
  build_g_kernel<<<(5 * 4096) / 256, 256, 0, stream>>>(cw1, cw3, cw5, Gp);
  counts_kernel<<<BAT, 256, 0, stream>>>(mask, cnts);

  // QKV GEMMs: 2048x32 tiles, 8 waves/block, y-split over q/k/v
  qkv_kernel<<<dim3((BS / 16) * (DM / 16) / 8, 3), 256, 0, stream>>>(xh, Wp, Qh, Kh, Vh);

  // cluster gather + Linformer projections
  proj_kernel<<<BH, 256, 0, stream>>>(Kh, Vh, EW, FW, table, cnts, KpT, Vpj);

  // fused scores/conv/softmax/attn: 32 seq-tiles x 128 (b,h)
  fused_attn_kernel<<<dim3(SEQ / 64, BH), 256, 0, stream>>>(Qh, KpT, Vpj, Gp, Ch);

  // output dense projection
  dense_kernel<<<(BS / 16) * (DM / 16) / 8, 256, 0, stream>>>(
      Ch, Wp + (size_t)3 * DM * DM, db, out);
}
